// GatedDeltaNetScanFLA_4784593567888
// MI455X (gfx1250) — matrix operand order repaired and checked by reading, not compiled, not run on hardware
//
#include <hip/hip_runtime.h>
#include <hip/hip_bf16.h>
#include <math.h>

// Problem constants from the reference
#define Bq 2
#define Tq 2048
#define Dq 2048
#define Hq 16
#define Kq 128
#define Vq 128

typedef __attribute__((ext_vector_type(16))) __bf16 v16bf;
typedef __attribute__((ext_vector_type(8)))  float  v8f;
typedef __attribute__((ext_vector_type(4)))  unsigned v4u;
typedef __attribute__((ext_vector_type(8)))  int   v8i;
typedef __attribute__((ext_vector_type(4)))  int   v4i;

// ---- TDM availability (device pass only; host pass takes fallback) --------
#if defined(__AMDGCN__) && defined(__has_builtin)
# if __has_builtin(__builtin_amdgcn_tensor_load_to_lds) && __has_builtin(__builtin_amdgcn_s_wait_tensorcnt)
#  define USE_TDM 1
# endif
#endif
#ifndef USE_TDM
# define USE_TDM 0
#endif
#if USE_TDM && defined(__clang_major__) && (__clang_major__ >= 23)
# define TDM_6ARG 1
#else
# define TDM_6ARG 0
#endif

__device__ __forceinline__ __bf16 f2bf(float f) {
  union { float f; unsigned u; } c; c.f = f;
  unsigned r = c.u + 0x7FFFu + ((c.u >> 16) & 1u);   // round-to-nearest-even
  unsigned short hs = (unsigned short)(r >> 16);
  __bf16 o; __builtin_memcpy(&o, &hs, 2);
  return o;
}

// fp32 -> bf16 buffer conversion (one-shot, HBM-bound, negligible)
__global__ __launch_bounds__(256) void to_bf16_kernel(
    const float* __restrict__ in, __bf16* __restrict__ out, int n)
{
  const int i = (blockIdx.x * 256 + threadIdx.x) * 4;
  if (i < n) {
    #pragma unroll
    for (int j = 0; j < 4; ++j) out[i + j] = f2bf(in[i + j]);
  }
}

// ---------------------------------------------------------------------------
// WMMA GEMM: Y[M][N] = Xb[M][Kd] * Wb[N][Kd]^T  (bf16 in, fp32 out)
// Block: 256 threads = 8 waves. Block tile 128(M) x 64(N), K-step 64.
// Wave grid 4(M) x 2(N): each wave owns 32x32 = 2x2 WMMA tiles ->
// 8 x v_wmma_f32_16x16x32_bf16 per K-stage per wave.
// Staging: TDM tensor_load_to_lds, double buffered, TENSORcnt-pipelined.
// TDM pad (1 DWORD per 32 DWORDs) gives a 132B LDS row stride (bank-friendly).
// ---------------------------------------------------------------------------
#define GM 128
#define GN 64
#define GK 64
#define LDS_ROW_B 132                       // 64*2 + 4 pad bytes
#define LDS_A_BYTES (GM * LDS_ROW_B)        // 16896
#define LDS_B_BYTES (GN * LDS_ROW_B)        // 8448
#define LDS_BUF_BYTES (LDS_A_BYTES + LDS_B_BYTES)

#if USE_TDM
// Issue one 2D bf16 tile load (tile_k x tile_rows, row stride in elements)
__device__ __forceinline__ void tdm_load_2d_bf16(
    unsigned lds_byte_addr, const __bf16* gsrc,
    unsigned tile_k, unsigned tile_rows, unsigned row_stride_elems)
{
  const unsigned long long ga = (unsigned long long)(size_t)gsrc;
  v4u g0;
  g0[0] = 1u;                                        // count=1 (valid D#)
  g0[1] = lds_byte_addr;                             // lds_addr bits[63:32]
  g0[2] = (unsigned)(ga & 0xFFFFFFFFu);              // global_addr[31:0]
  g0[3] = (unsigned)((ga >> 32) & 0x01FFFFFFu)       // global_addr[56:32]
        | (2u << 30);                                // type=2 ("image")
  v8i g1;
  // data_size=1 (2B) | pad_enable | pad_interval=32DW (code 4) | pad_amount=1DW (code 0)
  const unsigned w0 = (1u << 16) | (1u << 20) | (4u << 22);
  const unsigned td0 = row_stride_elems;             // tensor_dim0 (never OOB)
  const unsigned td1 = 0x00100000u;                  // tensor_dim1: large
  g1[0] = (int)w0;
  g1[1] = (int)((td0 & 0xFFFFu) << 16);                              // td0[15:0]
  g1[2] = (int)(((td0 >> 16) & 0xFFFFu) | ((td1 & 0xFFFFu) << 16));  // td0[31:16], td1[15:0]
  g1[3] = (int)(((td1 >> 16) & 0xFFFFu) | ((tile_k & 0xFFFFu) << 16)); // td1[31:16], tile_dim0
  g1[4] = (int)(tile_rows & 0xFFFFu);                // tile_dim1 ; tile_dim2=0
  g1[5] = (int)row_stride_elems;                     // tensor_dim0_stride[31:0]
  g1[6] = 0;                                         // stride[47:32] | dim1_stride lo
  g1[7] = 0;
  const v4i z = {0, 0, 0, 0};
#if TDM_6ARG
  const v8i z8 = {0, 0, 0, 0, 0, 0, 0, 0};
  __builtin_amdgcn_tensor_load_to_lds(g0, g1, z, z, z8, 0);
#else
  __builtin_amdgcn_tensor_load_to_lds(g0, g1, z, z, 0);
#endif
}
#endif // USE_TDM

__device__ __forceinline__ v8f wmma_bf16(v16bf a, v16bf b, v8f c) {
  return __builtin_amdgcn_wmma_f32_16x16x32_bf16(false, a, false, b,
                                                 (short)0, c, false, false);
}

// One K-stage: 8 WMMA per wave (2x2 tiles x 2 K-halves)
__device__ __forceinline__ void gemm_stage(
    const unsigned char* Ab, const unsigned char* Bb,
    int rowA0, int rowA1, int colB0, int colB1, int hi, v8f acc[2][2])
{
  #pragma unroll
  for (int khf = 0; khf < 2; ++khf) {
    v16bf a0, a1, b0, b1;
    #pragma unroll
    for (int e = 0; e < 16; ++e) {
      const int ka = khf * 32 + e + ((e >= 8) ? 8 : 0) + (hi ? 8 : 0);
      a0[e] = *(const __bf16*)(Ab + rowA0 * LDS_ROW_B + ka * 2);
      a1[e] = *(const __bf16*)(Ab + rowA1 * LDS_ROW_B + ka * 2);
      const int kb = khf * 32 + e + ((e >= 8) ? 8 : 0) + (hi ? 8 : 0);
      b0[e] = *(const __bf16*)(Bb + colB0 * LDS_ROW_B + kb * 2);
      b1[e] = *(const __bf16*)(Bb + colB1 * LDS_ROW_B + kb * 2);
    }
    acc[0][0] = wmma_bf16(a0, b0, acc[0][0]);
    acc[0][1] = wmma_bf16(a0, b1, acc[0][1]);
    acc[1][0] = wmma_bf16(a1, b0, acc[1][0]);
    acc[1][1] = wmma_bf16(a1, b1, acc[1][1]);
  }
}

__global__ __launch_bounds__(256) void wmma_gemm_bf16(
    const __bf16* __restrict__ X, const __bf16* __restrict__ W,
    float* __restrict__ Y, int M, int N, int Kd)
{
  __shared__ __align__(16) unsigned char smem[2 * LDS_BUF_BYTES];

  const int tid = threadIdx.x;
  const int m0 = blockIdx.y * GM;
  const int n0 = blockIdx.x * GN;

  const int w     = tid >> 5;
  const int l     = tid & 31;
  const int waveM = (w & 3) * 32;   // 4 waves along M
  const int waveN = (w >> 2) * 32;  // 2 waves along N
  const int hi    = l >> 4;
  const int l15   = l & 15;

  const int rowA0 = waveM + l15;
  const int rowA1 = waveM + 16 + l15;
  const int colB0 = waveN + l15;
  const int colB1 = waveN + 16 + l15;

  v8f acc[2][2] = {};

  const int nst = Kd / GK;

#if USE_TDM
  if (tid < 32) {   // wave 0 drives the Tensor Data Mover
    tdm_load_2d_bf16((unsigned)(size_t)&smem[0],
                     X + (size_t)m0 * Kd, GK, GM, (unsigned)Kd);
    tdm_load_2d_bf16((unsigned)(size_t)&smem[LDS_A_BYTES],
                     W + (size_t)n0 * Kd, GK, GN, (unsigned)Kd);
  }
  for (int s = 0; s < nst; ++s) {
    const int cur = s & 1;
    if (tid < 32) {
      if (s + 1 < nst) {
        const int k1 = (s + 1) * GK;
        const unsigned nb = (unsigned)(cur ^ 1) * LDS_BUF_BYTES;
        tdm_load_2d_bf16((unsigned)(size_t)&smem[nb],
                         X + (size_t)m0 * Kd + k1, GK, GM, (unsigned)Kd);
        tdm_load_2d_bf16((unsigned)(size_t)&smem[nb + LDS_A_BYTES],
                         W + (size_t)n0 * Kd + k1, GK, GN, (unsigned)Kd);
        __builtin_amdgcn_s_wait_tensorcnt(2);   // current buffer retired
      } else {
        __builtin_amdgcn_s_wait_tensorcnt(0);
      }
    }
    __syncthreads();
    const unsigned char* base = smem + (unsigned)cur * LDS_BUF_BYTES;
    gemm_stage(base, base + LDS_A_BYTES, rowA0, rowA1, colB0, colB1, hi, acc);
    __syncthreads();
  }
#else
  // Fallback: manual staged copies into the same padded LDS layout
  for (int s = 0; s < nst; ++s) {
    const int k0 = s * GK;
    {
      const int r  = tid >> 1;            // 0..127
      const int c0 = (tid & 1) * 32;
      const __bf16* src = X + (size_t)(m0 + r) * Kd + k0 + c0;
      unsigned char* dst = smem + r * LDS_ROW_B + c0 * 2;
      #pragma unroll
      for (int j = 0; j < 32; ++j) *(__bf16*)(dst + j * 2) = src[j];
    }
    {
      const int r  = tid >> 2;            // 0..63
      const int c0 = (tid & 3) * 16;
      const __bf16* src = W + (size_t)(n0 + r) * Kd + k0 + c0;
      unsigned char* dst = smem + LDS_A_BYTES + r * LDS_ROW_B + c0 * 2;
      #pragma unroll
      for (int j = 0; j < 16; ++j) *(__bf16*)(dst + j * 2) = src[j];
    }
    __syncthreads();
    gemm_stage(smem, smem + LDS_A_BYTES, rowA0, rowA1, colB0, colB1, hi, acc);
    __syncthreads();
  }
#endif

  // C/D layout: VGPR r -> M = r + (hi?8:0), N = l&15
  #pragma unroll
  for (int mi = 0; mi < 2; ++mi) {
    #pragma unroll
    for (int ni = 0; ni < 2; ++ni) {
      #pragma unroll
      for (int r = 0; r < 8; ++r) {
        const int m = m0 + waveM + mi * 16 + r + (hi ? 8 : 0);
        const int n = n0 + waveN + ni * 16 + l15;
        Y[(size_t)m * N + n] = acc[mi][ni][r];
      }
    }
  }
}

// ---------------------------------------------------------------------------
// Gate projections: alpha = sigmoid(x.Wa^T + ba), beta = softplus(x.Wb^T+bb)*K^-0.5
// ---------------------------------------------------------------------------
__global__ __launch_bounds__(256) void proj_gates(
    const float* __restrict__ x,
    const float* __restrict__ Wa, const float* __restrict__ ba,
    const float* __restrict__ Wb, const float* __restrict__ bb,
    float* __restrict__ alpha, float* __restrict__ beta)
{
  const int token = blockIdx.x;
  const int tid = threadIdx.x;
  const int out = tid >> 3;        // 0..31 (16 alpha + 16 beta)
  const int sub = tid & 7;
  const int h = out & 15;
  const bool isB = out >= 16;
  const float* wrow = (isB ? Wb : Wa) + (size_t)h * Dq;
  const float* xrow = x + (size_t)token * Dq;
  float s = 0.f;
  const int c0 = sub * (Dq / 8);
  for (int j = 0; j < Dq / 8; ++j) s = fmaf(xrow[c0 + j], wrow[c0 + j], s);
  s += __shfl_xor(s, 1, 32);
  s += __shfl_xor(s, 2, 32);
  s += __shfl_xor(s, 4, 32);
  if (sub == 0) {
    if (!isB) {
      const float z = s + ba[h];
      alpha[(size_t)token * Hq + h] = 1.0f / (1.0f + expf(-z));
    } else {
      const float z = s + bb[h];
      const float sp = (z > 20.0f) ? z : log1pf(expf(z));
      beta[(size_t)token * Hq + h] = sp * 0.08838834764831845f;  // K^-0.5
    }
  }
}

// ---------------------------------------------------------------------------
// Per-head L2 normalization: one wave32 per (token, head), 128 values
// ---------------------------------------------------------------------------
__global__ __launch_bounds__(32) void l2norm_heads(float* __restrict__ buf)
{
  const int bid = blockIdx.x;               // token*H + h
  const int token = bid >> 4, h = bid & 15;
  float* p = buf + (size_t)token * (Hq * Kq) + h * Kq + threadIdx.x;
  float v0 = p[0], v1 = p[32], v2 = p[64], v3 = p[96];
  float ss = v0 * v0 + v1 * v1 + v2 * v2 + v3 * v3;
  #pragma unroll
  for (int m = 16; m >= 1; m >>= 1) ss += __shfl_xor(ss, m, 32);
  const float scale = 1.0f / fmaxf(sqrtf(ss), 1e-12f);
  p[0] = v0 * scale; p[32] = v1 * scale; p[64] = v2 * scale; p[96] = v3 * scale;
}

// ---------------------------------------------------------------------------
// Gated delta-rule scan. One block per (b,h); S[128x128] fp32 in registers.
// ---------------------------------------------------------------------------
__global__ __launch_bounds__(256) void gdn_scan(
    const float* __restrict__ qb, const float* __restrict__ kb,
    const float* __restrict__ vb,
    const float* __restrict__ al, const float* __restrict__ be,
    float* __restrict__ out)
{
  const int bh = blockIdx.x;
  const int b = bh >> 4, h = bh & 15;
  const int tid = threadIdx.x;
  const int vrow = tid >> 1;
  const int kh = (tid & 1) * 64;

  __shared__ float ks[128];
  __shared__ float qs[128];

  float S[64];
  #pragma unroll
  for (int j = 0; j < 64; ++j) S[j] = 0.f;

  for (int t = 0; t < Tq; ++t) {
    const size_t tok = (size_t)b * Tq + t;
    const size_t off = tok * (Hq * Kq) + (size_t)h * Kq;

    if (t + 8 < Tq) {   // global_prefetch_b8: keep the stream hot
      const size_t poff = (tok + 8) * (Hq * Kq) + (size_t)h * Kq;
      __builtin_prefetch(kb + poff + (tid & 127), 0, 1);
      __builtin_prefetch(qb + poff + (tid & 127), 0, 1);
    }

    if (tid < 128) ks[tid] = kb[off + tid];
    else           qs[tid - 128] = qb[off + (tid - 128)];
    const float vval = vb[off + vrow];
    const float a  = al[tok * Hq + h];
    const float bt = be[tok * Hq + h];
    __syncthreads();

    float dot = 0.f;
    #pragma unroll
    for (int j = 0; j < 64; ++j) dot = fmaf(S[j], ks[kh + j], dot);
    dot += __shfl_xor(dot, 1, 32);
    const float u = bt * (vval - a * dot);   // pred = a * (S.k)

    float od = 0.f;
    #pragma unroll
    for (int j = 0; j < 64; ++j) {
      const float kv = ks[kh + j];
      S[j] = fmaf(a, S[j], u * kv);          // S = a*S + u k^T
      od = fmaf(S[j], qs[kh + j], od);       // o = S q
    }
    od += __shfl_xor(od, 1, 32);
    if ((tid & 1) == 0)
      out[(tok * Hq + h) * Vq + vrow] = od;
    __syncthreads();
  }
}

// ---------------------------------------------------------------------------
extern "C" void kernel_launch(void* const* d_in, const int* in_sizes, int n_in,
                              void* d_out, int out_size, void* d_ws, size_t ws_size,
                              hipStream_t stream)
{
  const float* x  = (const float*)d_in[0];
  const float* Wq = (const float*)d_in[1];
  const float* Wk = (const float*)d_in[2];
  const float* Wv = (const float*)d_in[3];
  const float* Wa = (const float*)d_in[4];
  const float* ba = (const float*)d_in[5];
  const float* Wb = (const float*)d_in[6];
  const float* bb = (const float*)d_in[7];
  float* out = (float*)d_out;

  char* ws = (char*)d_ws;
  const size_t MTOK = (size_t)Bq * Tq;           // 4096 tokens
  const size_t MN   = MTOK * Hq * Kq;            // 4096 x 2048
  const size_t WN   = (size_t)Hq * Kq * Dq;      // 2048 x 2048
  float*  qbuf  = (float*)ws;  ws += MN * sizeof(float);
  float*  kbuf  = (float*)ws;  ws += MN * sizeof(float);
  float*  vbuf  = (float*)ws;  ws += MN * sizeof(float);
  float*  albuf = (float*)ws;  ws += MTOK * Hq * sizeof(float);
  float*  bebuf = (float*)ws;  ws += MTOK * Hq * sizeof(float);
  __bf16* xb    = (__bf16*)ws; ws += MN * sizeof(__bf16);
  __bf16* wqb   = (__bf16*)ws; ws += WN * sizeof(__bf16);
  __bf16* wkb   = (__bf16*)ws; ws += WN * sizeof(__bf16);
  __bf16* wvb   = (__bf16*)ws; ws += WN * sizeof(__bf16);

  // fp32 -> bf16 once (HBM-bound)
  to_bf16_kernel<<<(int)(MN / 1024), 256, 0, stream>>>(x,  xb,  (int)MN);
  to_bf16_kernel<<<(int)(WN / 1024), 256, 0, stream>>>(Wq, wqb, (int)WN);
  to_bf16_kernel<<<(int)(WN / 1024), 256, 0, stream>>>(Wk, wkb, (int)WN);
  to_bf16_kernel<<<(int)(WN / 1024), 256, 0, stream>>>(Wv, wvb, (int)WN);

  const dim3 g((Hq * Kq) / GN, (Bq * Tq) / GM);  // (32, 32)
  wmma_gemm_bf16<<<g, 256, 0, stream>>>(xb, wqb, qbuf, Bq * Tq, Hq * Kq, Dq);
  wmma_gemm_bf16<<<g, 256, 0, stream>>>(xb, wkb, kbuf, Bq * Tq, Hq * Kq, Dq);
  wmma_gemm_bf16<<<g, 256, 0, stream>>>(xb, wvb, vbuf, Bq * Tq, Hq * Vq, Dq);
  proj_gates<<<Bq * Tq, 256, 0, stream>>>(x, Wa, ba, Wb, bb, albuf, bebuf);
  l2norm_heads<<<Bq * Tq * Hq, 32, 0, stream>>>(qbuf);
  l2norm_heads<<<Bq * Tq * Hq, 32, 0, stream>>>(kbuf);
  gdn_scan<<<Bq * Hq, 256, 0, stream>>>(qbuf, kbuf, vbuf, albuf, bebuf, out);
}
